// GATClassifier_44684839747645
// MI455X (gfx1250) — matrix operand order repaired and checked by reading, not compiled, not run on hardware
//
#include <hip/hip_runtime.h>
#include <hip/hip_bf16.h>

#define N_NODES  50000
#define N_EDGES  800000
#define N_GRAPHS 500

typedef __attribute__((ext_vector_type(16))) __bf16 bf16x16;
typedef __attribute__((ext_vector_type(8)))  __bf16 bf16x8;
typedef __attribute__((ext_vector_type(8)))  float  f32x8;

static __device__ __forceinline__ __bf16 f2bf(float x) {
  union { float f; unsigned u; } v; v.f = x;
  unsigned r = v.u + 0x7FFFu + ((v.u >> 16) & 1u);   // round-to-nearest-even
  unsigned short h = (unsigned short)(r >> 16);
  __bf16 o; __builtin_memcpy(&o, &h, sizeof(o));
  return o;
}

// ---------------------------------------------------------------- utilities
__global__ void fill_f32(float* __restrict__ p, float v, int n) {
  int i = blockIdx.x * blockDim.x + threadIdx.x;
  if (i < n) p[i] = v;
}

__global__ void convert_bf16(const float* __restrict__ w, __bf16* __restrict__ o, int n) {
  int i = blockIdx.x * blockDim.x + threadIdx.x;
  if (i < n) o[i] = f2bf(w[i]);
}

static __device__ __forceinline__ float atomicMaxF32(float* addr, float val) {
  int* ia = (int*)addr;
  int old = __hip_atomic_load(ia, __ATOMIC_RELAXED, __HIP_MEMORY_SCOPE_AGENT);
  while (__int_as_float(old) < val) {
    int assumed = old;
    old = atomicCAS(ia, assumed, __float_as_int(val));
    if (old == assumed) break;
  }
  return __int_as_float(old);
}

// ------------------------------------------------ weight pre-pack into B lane layout
// B 32x16 bf16 WMMA layout: lane holds col = nT*16 + (lane&15); its 16 elements are
// K = kb + 8*(lane>>4) + e (e<8), kb + 16 + 8*(lane>>4) + e-8 (e>=8): two 8-elem K runs. Pack so each lane's operand
// is one contiguous 32-byte chunk: wbp[((blk*nTiles + nT)*32 + lane)*16 + e].
template<int FOUT>
__global__ void pack_w(const float* __restrict__ W, __bf16* __restrict__ o) {
  int i = blockIdx.x * blockDim.x + threadIdx.x;        // 128*FOUT total
  if (i >= 128 * FOUT) return;
  const int nTiles = FOUT / 16;
  int e    = i & 15;
  int lane = (i >> 4) & 31;
  int t    = i >> 9;                                    // blk*nTiles + nT
  int blk  = t / nTiles, nT = t % nTiles;
  int col  = nT * 16 + (lane & 15);
  int K    = blk * 32 + (e < 8 ? 8 * (lane >> 4) + e : 16 + 8 * (lane >> 4) + (e - 8));
  o[i] = f2bf(W[K * FOUT + col]);
}

// -------------------------------------------------- WMMA GEMM: C = A(bf16) x Bp(bf16)
// A: [M,128] bf16 row-major (the ISA A 16x32 lane layout == two contiguous 8-elem runs
// per lane -> two global_load_b128). Bp: pre-packed lane-major. C: [M,FOUT] fp32.
// One wave32 computes one 16x16 tile; K=128 in four v_wmma_f32_16x16x32_bf16 steps.
template<int FOUT>
__global__ void gemm_wmma_bf16(const __bf16* __restrict__ A, const __bf16* __restrict__ Bp,
                               float* __restrict__ C, int M) {
  const int lane = threadIdx.x & 31;
  const int wave = (blockIdx.x * blockDim.x + threadIdx.x) >> 5;
  const int nTiles = FOUT / 16;
  const int totalTiles = (M / 16) * nTiles;
  if (wave >= totalTiles) return;          // uniform per wave -> EXEC all-ones at WMMA
  const int mTile = wave / nTiles;
  const int nTile = wave % nTiles;
  const int r    = lane & 15;
  const int half = lane >> 4;

  const __bf16* arow = A + (size_t)(mTile * 16 + r) * 128 + 8 * half;
  f32x8 acc = {};
  #pragma unroll
  for (int kb = 0; kb < 128; kb += 32) {
    union { bf16x16 v; bf16x8 h[2]; } a;
    a.h[0] = *(const bf16x8*)(arow + kb);        // K = kb +    8*half .. +7
    a.h[1] = *(const bf16x8*)(arow + kb + 16);   // K = kb+16 + 8*half .. +7
    const bf16x16 b =
        *(const bf16x16*)(Bp + (((size_t)(kb >> 5) * nTiles + nTile) * 32 + lane) * 16);
    acc = __builtin_amdgcn_wmma_f32_16x16x32_bf16(false, a.v, false, b,
                                                  (short)0, acc, false, false);
  }
  // C/D layout: VGPR rr -> row rr (lanes 0..15) / rr+8 (lanes 16..31), col = lane&15
  #pragma unroll
  for (int rr = 0; rr < 8; ++rr) {
    const int mrow = mTile * 16 + rr + 8 * half;
    C[(size_t)mrow * FOUT + nTile * 16 + r] = acc[rr];
  }
}

// ------------------------------------------------------------- GAT edge stage
__global__ void scores(const float* __restrict__ feat, const float* __restrict__ al,
                       const float* __restrict__ ar, float* __restrict__ el,
                       float* __restrict__ er, int N, int H, int D) {
  int idx = blockIdx.x * blockDim.x + threadIdx.x;       // n*H + h
  if (idx >= N * H) return;
  int h = idx % H;
  const float* f = feat + (size_t)idx * D;
  float sl = 0.f, sr = 0.f;
  for (int d = 0; d < D; ++d) { float v = f[d]; sl += v * al[h * D + d]; sr += v * ar[h * D + d]; }
  el[idx] = sl; er[idx] = sr;
}

__global__ void edge_max(const int* __restrict__ src, const int* __restrict__ dst,
                         const float* __restrict__ el, const float* __restrict__ er,
                         float* __restrict__ mmax, int E, int H) {
  int idx = blockIdx.x * blockDim.x + threadIdx.x;       // e*H + h
  if (idx >= E * H) return;
  int e = idx / H, h = idx % H;
  int s = src[e], d = dst[e];
  float v = el[s * H + h] + er[d * H + h];
  v = v > 0.f ? v : 0.2f * v;
  atomicMaxF32(&mmax[d * H + h], v);
}

__global__ void edge_expsum(const int* __restrict__ src, const int* __restrict__ dst,
                            const float* __restrict__ el, const float* __restrict__ er,
                            const float* __restrict__ mmax, float* __restrict__ den,
                            int E, int H) {
  int idx = blockIdx.x * blockDim.x + threadIdx.x;
  if (idx >= E * H) return;
  int e = idx / H, h = idx % H;
  int s = src[e], d = dst[e];
  float v = el[s * H + h] + er[d * H + h];
  v = v > 0.f ? v : 0.2f * v;
  atomicAdd(&den[d * H + h], __expf(v - mmax[d * H + h]));
}

// one wave per edge; lanes cover D channels; alpha recomputed (cheaper than a 12.8MB
// per-edge buffer round trip at 23.3TB/s)
__global__ void edge_aggregate(const int* __restrict__ src, const int* __restrict__ dst,
                               const float* __restrict__ el, const float* __restrict__ er,
                               const float* __restrict__ mmax, const float* __restrict__ den,
                               const float* __restrict__ feat, float* __restrict__ agg,
                               int E, int H, int D) {
  int wid  = (blockIdx.x * blockDim.x + threadIdx.x) >> 5;
  int lane = threadIdx.x & 31;
  if (wid >= E) return;
  int s = src[wid], d = dst[wid];
  for (int h = 0; h < H; ++h) {
    float v = el[s * H + h] + er[d * H + h];
    v = v > 0.f ? v : 0.2f * v;
    float alpha = __expf(v - mmax[d * H + h]) / den[d * H + h];
    for (int c = lane; c < D; c += 32) {
      atomicAdd(&agg[((size_t)d * H + h) * D + c],
                feat[((size_t)s * H + h) * D + c] * alpha);
    }
  }
}

// -------------------------------------------------------------- BatchNorm+ReLU
__global__ void bn_stats(const float* __restrict__ agg, const float* __restrict__ bias,
                         float* __restrict__ stats, int N, int C) {
  int c = threadIdx.x;                       // blockDim.x == C
  float s = 0.f, s2 = 0.f;
  for (int n = blockIdx.x; n < N; n += gridDim.x) {
    float v = agg[(size_t)n * C + c] + bias[c];
    s += v; s2 += v * v;
  }
  atomicAdd(&stats[c], s);
  atomicAdd(&stats[C + c], s2);
}

// writes bf16 directly: the BN output is consumed only by the next layer's GEMM
__global__ void bn_apply(const float* __restrict__ agg, const float* __restrict__ bias,
                         const float* __restrict__ stats, const float* __restrict__ gamma,
                         const float* __restrict__ beta, __bf16* __restrict__ out,
                         int N, int C) {
  int i = blockIdx.x * blockDim.x + threadIdx.x;
  if (i >= N * C) return;
  int c = i % C;
  float invN = 1.0f / (float)N;
  float mu   = stats[c] * invN;
  float var  = stats[C + c] * invN - mu * mu;
  float v = (agg[i] + bias[c] - mu) * __frsqrt_rn(var + 1e-5f) * gamma[c] + beta[c];
  out[i] = f2bf(v > 0.f ? v : 0.f);
}

// ------------------------------------------------------------------ pool + FC
__global__ void pool(const float* __restrict__ h2, const float* __restrict__ b2,
                     const int* __restrict__ gid, float* __restrict__ hg, int N, int C) {
  int i = blockIdx.x * blockDim.x + threadIdx.x;
  if (i >= N * C) return;
  int n = i / C, c = i % C;
  atomicAdd(&hg[gid[n] * C + c], h2[i] + b2[c]);
}

__global__ void classify(const float* __restrict__ hg, const float* __restrict__ Wc,
                         const float* __restrict__ bc, float* __restrict__ out, int G) {
  int idx = blockIdx.x * blockDim.x + threadIdx.x;       // g*10 + k
  if (idx >= G * 10) return;
  int g = idx / 10, k = idx % 10;
  float s = bc[k];
  for (int c = 0; c < 32; ++c) s += hg[g * 32 + c] * Wc[c * 10 + k];
  out[idx] = s;
}

// ---------------------------------------------------------------------- driver
static inline int cdiv(long long a, long long b) { return (int)((a + b - 1) / b); }

extern "C" void kernel_launch(void* const* d_in, const int* in_sizes, int n_in,
                              void* d_out, int out_size, void* d_ws, size_t ws_size,
                              hipStream_t stream) {
  const float* x    = (const float*)d_in[0];
  const int*   esrc = (const int*)d_in[1];
  const int*   edst = (const int*)d_in[2];
  const int*   gid  = (const int*)d_in[3];
  const float* W0   = (const float*)d_in[4];
  const float* al0  = (const float*)d_in[5];
  const float* ar0  = (const float*)d_in[6];
  const float* b0   = (const float*)d_in[7];
  const float* W1   = (const float*)d_in[8];
  const float* al1  = (const float*)d_in[9];
  const float* ar1  = (const float*)d_in[10];
  const float* b1   = (const float*)d_in[11];
  const float* W2   = (const float*)d_in[12];
  const float* al2  = (const float*)d_in[13];
  const float* ar2  = (const float*)d_in[14];
  const float* b2   = (const float*)d_in[15];
  const float* g0   = (const float*)d_in[16];
  const float* be0  = (const float*)d_in[17];
  const float* g1   = (const float*)d_in[18];
  const float* be1  = (const float*)d_in[19];
  const float* Wc   = (const float*)d_in[20];
  const float* bc   = (const float*)d_in[21];
  float* out = (float*)d_out;

  // workspace layout (all vector-load regions 32B-aligned: offsets are multiples of 32)
  char*   wsb   = (char*)d_ws;
  __bf16* abf   = (__bf16*)wsb;                                   // N*128 bf16 (12.8MB)
  __bf16* wbp   = (__bf16*)(wsb + (size_t)N_NODES * 128 * 2);     // 128*128 bf16 packed
  float*  feat  = (float*)(wsb + (size_t)N_NODES * 128 * 2 + 32768);   // N*128 f32
  float*  agg   = feat + (size_t)N_NODES * 128;                   // N*128 f32
  float*  el    = agg  + (size_t)N_NODES * 128;                   // N*4
  float*  er    = el   + (size_t)N_NODES * 4;
  float*  mmax  = er   + (size_t)N_NODES * 4;
  float*  den   = mmax + (size_t)N_NODES * 4;
  float*  stats = den  + (size_t)N_NODES * 4;                     // 256
  float*  hg    = stats + 256;                                    // G*32

  const int B = 256;
  const int H01 = 4, D01 = 32;

  // layer-0 GEMM input: x converted to bf16 once
  convert_bf16<<<cdiv((long long)N_NODES * 128, B), B, 0, stream>>>(x, abf, N_NODES * 128);

  const float* Ws[2]  = { W0, W1 };
  const float* als[2] = { al0, al1 };
  const float* ars[2] = { ar0, ar1 };
  const float* bs[2]  = { b0, b1 };
  const float* gs[2]  = { g0, g1 };
  const float* bes[2] = { be0, be1 };

  for (int L = 0; L < 2; ++L) {
    pack_w<128><<<cdiv(128 * 128, B), B, 0, stream>>>(Ws[L], wbp);
    { int tiles = (N_NODES / 16) * (128 / 16);
      gemm_wmma_bf16<128><<<cdiv((long long)tiles * 32, B), B, 0, stream>>>(abf, wbp, feat, N_NODES); }
    scores<<<cdiv(N_NODES * H01, B), B, 0, stream>>>(feat, als[L], ars[L], el, er, N_NODES, H01, D01);
    fill_f32<<<cdiv(N_NODES * H01, B), B, 0, stream>>>(mmax, -1e30f, N_NODES * H01);
    fill_f32<<<cdiv(N_NODES * H01, B), B, 0, stream>>>(den, 0.f, N_NODES * H01);
    fill_f32<<<cdiv(N_NODES * 128, B), B, 0, stream>>>(agg, 0.f, N_NODES * 128);
    edge_max<<<cdiv(N_EDGES * H01, B), B, 0, stream>>>(esrc, edst, el, er, mmax, N_EDGES, H01);
    edge_expsum<<<cdiv(N_EDGES * H01, B), B, 0, stream>>>(esrc, edst, el, er, mmax, den, N_EDGES, H01);
    edge_aggregate<<<cdiv((long long)N_EDGES * 32, B), B, 0, stream>>>(
        esrc, edst, el, er, mmax, den, feat, agg, N_EDGES, H01, D01);
    fill_f32<<<1, 256, 0, stream>>>(stats, 0.f, 256);
    bn_stats<<<512, 128, 0, stream>>>(agg, bs[L], stats, N_NODES, 128);
    bn_apply<<<cdiv(N_NODES * 128, B), B, 0, stream>>>(agg, bs[L], stats, gs[L], bes[L], abf,
                                                       N_NODES, 128);
  }

  // ======================= layer 2 (H=1, D=32) =======================
  pack_w<32><<<cdiv(128 * 32, B), B, 0, stream>>>(W2, wbp);
  { int tiles = (N_NODES / 16) * (32 / 16);
    gemm_wmma_bf16<32><<<cdiv((long long)tiles * 32, B), B, 0, stream>>>(abf, wbp, feat, N_NODES); }
  scores<<<cdiv(N_NODES, B), B, 0, stream>>>(feat, al2, ar2, el, er, N_NODES, 1, 32);
  fill_f32<<<cdiv(N_NODES, B), B, 0, stream>>>(mmax, -1e30f, N_NODES);
  fill_f32<<<cdiv(N_NODES, B), B, 0, stream>>>(den, 0.f, N_NODES);
  fill_f32<<<cdiv(N_NODES * 32, B), B, 0, stream>>>(agg, 0.f, N_NODES * 32);
  edge_max<<<cdiv(N_EDGES, B), B, 0, stream>>>(esrc, edst, el, er, mmax, N_EDGES, 1);
  edge_expsum<<<cdiv(N_EDGES, B), B, 0, stream>>>(esrc, edst, el, er, mmax, den, N_EDGES, 1);
  edge_aggregate<<<cdiv((long long)N_EDGES * 32, B), B, 0, stream>>>(
      esrc, edst, el, er, mmax, den, feat, agg, N_EDGES, 1, 32);

  // ======================= pool + classifier =======================
  fill_f32<<<cdiv(N_GRAPHS * 32, B), B, 0, stream>>>(hg, 0.f, N_GRAPHS * 32);
  pool<<<cdiv(N_NODES * 32, B), B, 0, stream>>>(agg, b2, gid, hg, N_NODES, 32);
  classify<<<cdiv(N_GRAPHS * 10, B), B, 0, stream>>>(hg, Wc, bc, out, N_GRAPHS);
}
